// AblationAnomalyDetector_76562087019171
// MI455X (gfx1250) — matrix operand order repaired and checked by reading, not compiled, not run on hardware
//
#include <hip/hip_runtime.h>
#include <hip/hip_bf16.h>
#include <stdint.h>

#define NN    8192
#define DD    512
#define HEADS 8
#define HDIM  64
#define TOPK  16

typedef __bf16 bf16_t;
typedef bf16_t v16bf __attribute__((ext_vector_type(16)));
typedef float  v8f   __attribute__((ext_vector_type(8)));

typedef int v4i_t __attribute__((ext_vector_type(4)));
typedef __attribute__((address_space(1))) v4i_t g_v4i;   // global int4
typedef __attribute__((address_space(3))) v4i_t l_v4i;   // LDS int4

// ---------- scalar bf16 helpers ----------
__device__ inline bf16_t f2bf(float f) {
    uint32_t u = __builtin_bit_cast(uint32_t, f);
    u += 0x7fffu + ((u >> 16) & 1u);          // round to nearest even
    uint16_t h = (uint16_t)(u >> 16);
    return __builtin_bit_cast(bf16_t, h);
}
__device__ inline float bf2f(bf16_t x) {
    uint16_t h = __builtin_bit_cast(uint16_t, x);
    uint32_t u = ((uint32_t)h) << 16;
    return __builtin_bit_cast(float, u);
}

// ---------- CDNA5 async global->LDS copy (ASYNCcnt path) ----------
__device__ inline void async_copy_b128(const void* g, void* l) {
#if __has_builtin(__builtin_amdgcn_global_load_async_to_lds_b128)
    __builtin_amdgcn_global_load_async_to_lds_b128(
        (g_v4i*)(uintptr_t)g, (l_v4i*)l, 0, 0);
#else
    uint32_t lo = (uint32_t)(uintptr_t)(l_v4i*)l;
    asm volatile("global_load_async_to_lds_b128 %0, %1, off"
                 :: "v"(lo), "v"((unsigned long long)(uintptr_t)g) : "memory");
#endif
}
template <int N> __device__ inline void wait_asynccnt() {
#if __has_builtin(__builtin_amdgcn_s_wait_asynccnt)
    __builtin_amdgcn_s_wait_asynccnt(N);
#else
    asm volatile("s_wait_asynccnt %0" :: "i"(N) : "memory");
#endif
}

// ---------- WMMA fragment loaders (CDNA5 16x16x32 bf16 layouts) ----------
// A 16x32 (MxK): lane m (0-15) holds K 0..7 & 16..23 of row m; lane m+16 holds K 8..15 & 24..31.
__device__ inline v16bf load_a16x32(const bf16_t* __restrict__ A, int lda, int row0, int k0) {
    int lane = threadIdx.x & 31;
    int m  = lane & 15;
    int ks = lane >> 4;
    const bf16_t* p = A + (size_t)(row0 + m) * lda + k0 + ks * 8;
    union { v16bf v; uint4 q[2]; } u;
    u.q[0] = *(const uint4*)(p);
    u.q[1] = *(const uint4*)(p + 16);
    return u.v;
}
// B 32x16 (KxN) with B[k,n] = W[n,k] (W row-major [ncols, K]).
__device__ inline v16bf load_b32x16(const bf16_t* __restrict__ W, int ldw, int col0, int k0) {
    int lane = threadIdx.x & 31;
    int n  = lane & 15;
    int ks = lane >> 4;
    const bf16_t* p = W + (size_t)(col0 + n) * ldw + k0 + ks * 8;
    union { v16bf v; uint4 q[2]; } u;
    u.q[0] = *(const uint4*)(p);
    u.q[1] = *(const uint4*)(p + 16);
    return u.v;
}
// Same fragment shapes but sourced from an LDS tile with row stride 32 elems.
__device__ inline v16bf lds_a_frag(const bf16_t* tile, int lrow0) {
    int lane = threadIdx.x & 31;
    int m  = lane & 15;
    int ks = lane >> 4;
    const bf16_t* p = tile + (lrow0 + m) * 32 + ks * 8;
    union { v16bf v; uint4 q[2]; } u;
    u.q[0] = *(const uint4*)(p);
    u.q[1] = *(const uint4*)(p + 16);
    return u.v;
}
__device__ inline v16bf lds_b_frag(const bf16_t* tile, int lcol0) {
    int lane = threadIdx.x & 31;
    int n  = lane & 15;
    int ks = lane >> 4;
    const bf16_t* p = tile + (lcol0 + n) * 32 + ks * 8;
    union { v16bf v; uint4 q[2]; } u;
    u.q[0] = *(const uint4*)(p);
    u.q[1] = *(const uint4*)(p + 16);
    return u.v;
}
__device__ inline v8f wmma_bf16(v16bf a, v16bf b, v8f c) {
    return __builtin_amdgcn_wmma_f32_16x16x32_bf16(false, a, false, b, (short)0, c, false, false);
}

// ---------- f32 -> bf16 conversion ----------
__global__ void cvt_bf16_kernel(const float* __restrict__ in, bf16_t* __restrict__ out, int n) {
    int i = blockIdx.x * 256 + threadIdx.x;
    if (i < n) out[i] = f2bf(in[i]);
}

// ---------- per-modality: Linear -> ReLU -> LayerNorm, writes bf16 into seq[2n+tok] ----------
// GEMM phase software-pipelined like gemm_bias_kernel (prefetch next-k fragments).
__global__ __launch_bounds__(256) void proj_ln_kernel(
    const bf16_t* __restrict__ X, const bf16_t* __restrict__ W,
    const float* __restrict__ bias, const float* __restrict__ gamma,
    const float* __restrict__ beta, bf16_t* __restrict__ seq, int tok)
{
    __shared__ float tile[16][DD];
    __shared__ float red_s[16][16];
    __shared__ float red_q[16][16];
    __shared__ float mv[16][2];

    int row0 = blockIdx.x * 16;
    int wave = threadIdx.x >> 5;
    int lane = threadIdx.x & 31;
    int colw = wave * 64;

    v8f acc[4] = {};
    v16bf a = load_a16x32(X, DD, row0, 0);
    v16bf b[4];
    #pragma unroll
    for (int j = 0; j < 4; ++j) b[j] = load_b32x16(W, DD, colw + j * 16, 0);

    for (int kt = 0; kt < 16; ++kt) {
        v16bf na = a, nb[4] = { b[0], b[1], b[2], b[3] };
        if (kt < 15) {
            int kn = (kt + 1) * 32;
            na = load_a16x32(X, DD, row0, kn);
            #pragma unroll
            for (int j = 0; j < 4; ++j) nb[j] = load_b32x16(W, DD, colw + j * 16, kn);
        }
        #pragma unroll
        for (int j = 0; j < 4; ++j) acc[j] = wmma_bf16(a, b[j], acc[j]);
        a = na;
        #pragma unroll
        for (int j = 0; j < 4; ++j) b[j] = nb[j];
    }
    int cb = lane & 15, rsel = (lane >> 4) * 8;
    #pragma unroll
    for (int j = 0; j < 4; ++j) {
        int col = colw + j * 16 + cb;
        float bb = bias[col];
        #pragma unroll
        for (int i = 0; i < 8; ++i) {
            float v = acc[j][i] + bb;
            tile[i + rsel][col] = v > 0.f ? v : 0.f;
        }
    }
    __syncthreads();
    int r = threadIdx.x >> 4;
    int t = threadIdx.x & 15;
    float s = 0.f, q = 0.f;
    for (int c = t; c < DD; c += 16) { float v = tile[r][c]; s += v; q += v * v; }
    red_s[r][t] = s; red_q[r][t] = q;
    __syncthreads();
    if (t == 0) {
        float ss = 0.f, qq = 0.f;
        for (int i = 0; i < 16; ++i) { ss += red_s[r][i]; qq += red_q[r][i]; }
        float m   = ss * (1.f / DD);
        float var = qq * (1.f / DD) - m * m;
        mv[r][0] = m; mv[r][1] = rsqrtf(var + 1e-5f);
    }
    __syncthreads();
    float m = mv[r][0], rsd = mv[r][1];
    size_t orow = (size_t)(2 * (row0 + r) + tok) * DD;
    for (int c = t; c < DD; c += 16) {
        float v = (tile[r][c] - m) * rsd * gamma[c] + beta[c];
        seq[orow + c] = f2bf(v);
    }
}

// ---------- generic GEMM: out = scale*(A @ W^T) + bias, optional ReLU, bf16 out ----------
// Wave tile 32x64, software-pipelined fragment prefetch. grid.x = M/32.
__global__ __launch_bounds__(256) void gemm_bias_kernel(
    const bf16_t* __restrict__ A, const bf16_t* __restrict__ W,
    const float* __restrict__ bias, bf16_t* __restrict__ out,
    int ldo, float scale, int do_relu)
{
    int wave = threadIdx.x >> 5;
    int lane = threadIdx.x & 31;
    int row0 = blockIdx.x * 32;
    int col0 = blockIdx.y * ((blockDim.x >> 5) * 64) + wave * 64;

    v8f acc[2][4] = {};
    v16bf a0 = load_a16x32(A, DD, row0, 0);
    v16bf a1 = load_a16x32(A, DD, row0 + 16, 0);
    v16bf b[4];
    #pragma unroll
    for (int j = 0; j < 4; ++j) b[j] = load_b32x16(W, DD, col0 + j * 16, 0);

    for (int kt = 0; kt < 16; ++kt) {
        v16bf na0 = a0, na1 = a1, nb[4] = { b[0], b[1], b[2], b[3] };
        if (kt < 15) {
            int kn = (kt + 1) * 32;
            na0 = load_a16x32(A, DD, row0, kn);
            na1 = load_a16x32(A, DD, row0 + 16, kn);
            #pragma unroll
            for (int j = 0; j < 4; ++j) nb[j] = load_b32x16(W, DD, col0 + j * 16, kn);
        }
        #pragma unroll
        for (int j = 0; j < 4; ++j) {
            acc[0][j] = wmma_bf16(a0, b[j], acc[0][j]);
            acc[1][j] = wmma_bf16(a1, b[j], acc[1][j]);
        }
        a0 = na0; a1 = na1;
        #pragma unroll
        for (int j = 0; j < 4; ++j) b[j] = nb[j];
    }
    int cb = lane & 15, rsel = (lane >> 4) * 8;
    #pragma unroll
    for (int g = 0; g < 2; ++g) {
        #pragma unroll
        for (int j = 0; j < 4; ++j) {
            int col = col0 + j * 16 + cb;
            float bb = bias[col];
            #pragma unroll
            for (int i = 0; i < 8; ++i) {
                float v = acc[g][j][i] * scale + bb;
                if (do_relu) v = v > 0.f ? v : 0.f;
                out[(size_t)(row0 + g * 16 + i + rsel) * ldo + col] = f2bf(v);
            }
        }
    }
}

// ---------- tiny 2-token MHA per (node, head): one wave per task ----------
__global__ __launch_bounds__(256) void attn_kernel(const bf16_t* __restrict__ qkv,
                                                   bf16_t* __restrict__ ctxsum)
{
    int wave = threadIdx.x >> 5;
    int lane = threadIdx.x & 31;
    int task = blockIdx.x * 8 + wave;
    int n = task >> 3, h = task & 7;
    size_t r0 = (size_t)(2 * n) * (3 * DD);
    size_t r1 = r0 + 3 * DD;
    int base = h * HDIM;

    float q0[2], q1[2], k0v[2], k1v[2], v0[2], v1[2];
    #pragma unroll
    for (int j = 0; j < 2; ++j) {
        int d = base + lane + 32 * j;
        q0[j]  = bf2f(qkv[r0 + d]);             q1[j]  = bf2f(qkv[r1 + d]);
        k0v[j] = bf2f(qkv[r0 + DD + d]);        k1v[j] = bf2f(qkv[r1 + DD + d]);
        v0[j]  = bf2f(qkv[r0 + 2 * DD + d]);    v1[j]  = bf2f(qkv[r1 + 2 * DD + d]);
    }
    float s00 = q0[0] * k0v[0] + q0[1] * k0v[1];
    float s01 = q0[0] * k1v[0] + q0[1] * k1v[1];
    float s10 = q1[0] * k0v[0] + q1[1] * k0v[1];
    float s11 = q1[0] * k1v[0] + q1[1] * k1v[1];
    for (int off = 16; off; off >>= 1) {
        s00 += __shfl_xor(s00, off, 32);
        s01 += __shfl_xor(s01, off, 32);
        s10 += __shfl_xor(s10, off, 32);
        s11 += __shfl_xor(s11, off, 32);
    }
    const float sc = 0.125f;
    s00 *= sc; s01 *= sc; s10 *= sc; s11 *= sc;
    float m0 = fmaxf(s00, s01), e00 = __expf(s00 - m0), e01 = __expf(s01 - m0);
    float i0 = 1.f / (e00 + e01), a00 = e00 * i0, a01 = e01 * i0;
    float m1 = fmaxf(s10, s11), e10 = __expf(s10 - m1), e11 = __expf(s11 - m1);
    float i1 = 1.f / (e10 + e11), a10 = e10 * i1, a11 = e11 * i1;
    #pragma unroll
    for (int j = 0; j < 2; ++j) {
        float c0 = a00 * v0[j] + a01 * v1[j];
        float c1 = a10 * v0[j] + a11 * v1[j];
        ctxsum[(size_t)n * DD + base + lane + 32 * j] = f2bf(c0 + c1);
    }
}

// ---------- sim raw scores: S = F @ F^T via async-LDS double-buffered tiles ----------
// Block: 256 thr / 8 waves in 2x4; block tile 64 rows x 256 cols; K staged 32/step.
__global__ __launch_bounds__(256) void sim_kernel(const bf16_t* __restrict__ F,
                                                  float* __restrict__ S)
{
    __shared__ bf16_t Ash[2][64 * 32];    // 8 KB
    __shared__ bf16_t Bsh[2][256 * 32];   // 32 KB
    int tid  = threadIdx.x;
    int wave = tid >> 5;
    int lane = tid & 31;
    int row0 = blockIdx.x * 64;
    int colb = blockIdx.y * 256;
    int wr = (wave >> 2) * 32;            // wave row offset inside tile
    int wc = (wave & 3) * 64;             // wave col offset inside tile

    // stage tile kt (A: 64x32, B: 256x32) into LDS buffer `buf` (5 async b128/thread)
    auto stage = [&](int kt, int buf) {
        int arow = tid >> 2, ach = tid & 3;
        async_copy_b128(F + (size_t)(row0 + arow) * DD + kt * 32 + ach * 8,
                        &Ash[buf][arow * 32 + ach * 8]);
        #pragma unroll
        for (int c = 0; c < 4; ++c)
            async_copy_b128(F + (size_t)(colb + tid) * DD + kt * 32 + c * 8,
                            &Bsh[buf][tid * 32 + c * 8]);
    };

    stage(0, 0);
    v8f acc[2][4] = {};
    for (int kt = 0; kt < 16; ++kt) {
        int buf = kt & 1;
        if (kt < 15) {
            stage(kt + 1, buf ^ 1);
            wait_asynccnt<5>();           // tile kt done (async loads retire in order)
        } else {
            wait_asynccnt<0>();
        }
        __syncthreads();                  // tile kt visible to whole block
        v16bf af0 = lds_a_frag(Ash[buf], wr);
        v16bf af1 = lds_a_frag(Ash[buf], wr + 16);
        #pragma unroll
        for (int j = 0; j < 4; ++j) {
            v16bf bf = lds_b_frag(Bsh[buf], wc + j * 16);
            acc[0][j] = wmma_bf16(af0, bf, acc[0][j]);
            acc[1][j] = wmma_bf16(af1, bf, acc[1][j]);
        }
        __syncthreads();                  // done reading buf before it is re-staged
    }
    int cb = lane & 15, rsel = (lane >> 4) * 8;
    #pragma unroll
    for (int g = 0; g < 2; ++g) {
        #pragma unroll
        for (int j = 0; j < 4; ++j) {
            int col = colb + wc + j * 16 + cb;
            #pragma unroll
            for (int i = 0; i < 8; ++i)
                S[(size_t)(row0 + wr + g * 16 + i + rsel) * NN + col] = acc[g][j][i];
        }
    }
}

// ---------- per-row softmax stats + iterative top-16 (row cached in LDS) ----------
__global__ __launch_bounds__(256) void softmax_topk_kernel(
    const float* __restrict__ S, float* __restrict__ tv, int* __restrict__ ti)
{
    __shared__ float row[NN];
    __shared__ float rv[256];
    __shared__ int   ri[256];
    int tid = threadIdx.x;
    int r = blockIdx.x;
    const float* src = S + (size_t)r * NN;
    for (int c = tid; c < NN; c += 256) row[c] = src[c];
    __syncthreads();

    float lm = -1e30f;
    for (int c = tid; c < NN; c += 256) lm = fmaxf(lm, row[c]);
    rv[tid] = lm; __syncthreads();
    for (int s = 128; s > 0; s >>= 1) {
        if (tid < s) rv[tid] = fmaxf(rv[tid], rv[tid + s]);
        __syncthreads();
    }
    float mx = rv[0]; __syncthreads();

    float ls = 0.f;
    for (int c = tid; c < NN; c += 256) ls += __expf(row[c] - mx);
    rv[tid] = ls; __syncthreads();
    for (int s = 128; s > 0; s >>= 1) {
        if (tid < s) rv[tid] += rv[tid + s];
        __syncthreads();
    }
    float inv = 1.f / rv[0]; __syncthreads();

    for (int it = 0; it < TOPK; ++it) {
        float bv = -1e30f; int bi = 0;
        for (int c = tid; c < NN; c += 256) {
            float v = row[c];
            if (v > bv) { bv = v; bi = c; }
        }
        rv[tid] = bv; ri[tid] = bi; __syncthreads();
        for (int s = 128; s > 0; s >>= 1) {
            if (tid < s && rv[tid + s] > rv[tid]) { rv[tid] = rv[tid + s]; ri[tid] = ri[tid + s]; }
            __syncthreads();
        }
        if (tid == 0) {
            int idx = ri[0];
            tv[r * TOPK + it] = __expf(rv[0] - mx) * inv;
            ti[r * TOPK + it] = idx;
            row[idx] = -1e30f;
        }
        __syncthreads();
    }
}

// ---------- H build ----------
__global__ void zero4_kernel(float4* __restrict__ p, size_t n4) {
    size_t i = (size_t)blockIdx.x * blockDim.x + threadIdx.x;
    size_t stride = (size_t)gridDim.x * blockDim.x;
    float4 z = {0.f, 0.f, 0.f, 0.f};
    for (; i < n4; i += stride) p[i] = z;
}
__global__ void diag_kernel(float* __restrict__ H) {
    int i = blockIdx.x * 256 + threadIdx.x;
    if (i < NN) H[(size_t)i * NN + i] = 1.0f;
}
__global__ void scatter_kernel(float* __restrict__ H, const float* __restrict__ tv,
                               const int* __restrict__ ti) {
    int t = blockIdx.x * 256 + threadIdx.x;
    if (t < NN * TOPK) {
        int i = t >> 4;
        H[(size_t)ti[t] * NN + i] = tv[t];
    }
}
__global__ void colsum_kernel(const float* __restrict__ H, float* __restrict__ cs) {
    int c = blockIdx.x * 256 + threadIdx.x;
    float s = 0.f;
    for (int r = 0; r < NN; ++r) s += H[(size_t)r * NN + c];
    cs[c] = s;
}
__global__ void diagfix_kernel(float* __restrict__ H, const float* __restrict__ cs) {
    int i = blockIdx.x * 256 + threadIdx.x;
    if (i < NN && cs[i] == 0.f) H[(size_t)i * NN + i] = 1.0f;
}

// ---------- edge weights: sigmoid(hidden @ w2^T + b2) ----------
__global__ __launch_bounds__(256) void ew_kernel(const bf16_t* __restrict__ hidden,
                                                 const float* __restrict__ w2,
                                                 const float* __restrict__ b2,
                                                 float* __restrict__ out)
{
    int wave = threadIdx.x >> 5;
    int lane = threadIdx.x & 31;
    int n = blockIdx.x * 8 + wave;
    float s = 0.f;
    #pragma unroll
    for (int j = 0; j < 8; ++j) {
        int c = lane + 32 * j;
        s += bf2f(hidden[(size_t)n * 256 + c]) * w2[c];
    }
    for (int off = 16; off; off >>= 1) s += __shfl_xor(s, off, 32);
    if (lane == 0) {
        s += b2[0];
        float sg = 1.f / (1.f + __expf(-s));
        out[n] = fmaxf(sg, 1e-8f);
    }
}

// ---------- host launch ----------
extern "C" void kernel_launch(void* const* d_in, const int* in_sizes, int n_in,
                              void* d_out, int out_size, void* d_ws, size_t ws_size,
                              hipStream_t stream) {
    (void)in_sizes; (void)n_in; (void)out_size; (void)ws_size;
    const float* x0    = (const float*)d_in[0];
    const float* x1    = (const float*)d_in[1];
    const float* w_p0  = (const float*)d_in[2];
    const float* b_p0  = (const float*)d_in[3];
    const float* g0    = (const float*)d_in[4];
    const float* beta0 = (const float*)d_in[5];
    const float* w_p1  = (const float*)d_in[6];
    const float* b_p1  = (const float*)d_in[7];
    const float* g1    = (const float*)d_in[8];
    const float* beta1 = (const float*)d_in[9];
    const float* in_w  = (const float*)d_in[10];
    const float* in_b  = (const float*)d_in[11];
    const float* out_w = (const float*)d_in[12];
    const float* out_b = (const float*)d_in[13];
    const float* ew_w1 = (const float*)d_in[14];
    const float* ew_b1 = (const float*)d_in[15];
    const float* ew_w2 = (const float*)d_in[16];
    const float* ew_b2 = (const float*)d_in[17];

    float* H  = (float*)d_out;
    float* ew = (float*)d_out + (size_t)NN * NN;

    char* wp = (char*)d_ws;
    auto carve = [&](size_t bytes) { char* p = wp; wp += (bytes + 255) & ~(size_t)255; return p; };
    bf16_t* x0b   = (bf16_t*)carve((size_t)NN * DD * 2);
    bf16_t* x1b   = (bf16_t*)carve((size_t)NN * DD * 2);
    bf16_t* wp0b  = (bf16_t*)carve((size_t)DD * DD * 2);
    bf16_t* wp1b  = (bf16_t*)carve((size_t)DD * DD * 2);
    bf16_t* inwb  = (bf16_t*)carve((size_t)3 * DD * DD * 2);
    bf16_t* outwb = (bf16_t*)carve((size_t)DD * DD * 2);
    bf16_t* ew1b  = (bf16_t*)carve((size_t)256 * DD * 2);
    bf16_t* seqb  = (bf16_t*)carve((size_t)2 * NN * DD * 2);
    bf16_t* ctxb  = (bf16_t*)carve((size_t)NN * DD * 2);
    bf16_t* fusb  = (bf16_t*)carve((size_t)NN * DD * 2);
    bf16_t* hidb  = (bf16_t*)carve((size_t)NN * 256 * 2);
    float*  tv    = (float*)carve((size_t)NN * TOPK * 4);
    int*    ti    = (int*)carve((size_t)NN * TOPK * 4);
    float*  cs    = (float*)carve((size_t)NN * 4);

    bf16_t* qkvb = (bf16_t*)d_out;   // QKV scratch inside H region until sim overwrites it

    auto cvt = [&](const float* src, bf16_t* dst, int n) {
        cvt_bf16_kernel<<<(n + 255) / 256, 256, 0, stream>>>(src, dst, n);
    };
    cvt(x0, x0b, NN * DD);
    cvt(x1, x1b, NN * DD);
    cvt(w_p0, wp0b, DD * DD);
    cvt(w_p1, wp1b, DD * DD);
    cvt(in_w, inwb, 3 * DD * DD);
    cvt(out_w, outwb, DD * DD);
    cvt(ew_w1, ew1b, 256 * DD);

    proj_ln_kernel<<<NN / 16, 256, 0, stream>>>(x0b, wp0b, b_p0, g0, beta0, seqb, 0);
    proj_ln_kernel<<<NN / 16, 256, 0, stream>>>(x1b, wp1b, b_p1, g1, beta1, seqb, 1);

    // qkv = seq @ in_w^T + in_b, M = 2N rows, 1536 cols (3 col-blocks of 512)
    gemm_bias_kernel<<<dim3(2 * NN / 32, 3), 256, 0, stream>>>(
        seqb, inwb, in_b, qkvb, 3 * DD, 1.0f, 0);

    attn_kernel<<<NN * HEADS / 8, 256, 0, stream>>>(qkvb, ctxb);

    // fused = 0.5*(ctxsum @ out_w^T) + out_b
    gemm_bias_kernel<<<dim3(NN / 32, 1), 256, 0, stream>>>(
        ctxb, outwb, out_b, fusb, DD, 0.5f, 0);

    // raw similarity scores into d_out H region (async-LDS pipelined GEMM)
    sim_kernel<<<dim3(NN / 64, NN / 256), 256, 0, stream>>>(fusb, H);

    softmax_topk_kernel<<<NN, 256, 0, stream>>>(H, tv, ti);

    zero4_kernel<<<32768, 256, 0, stream>>>((float4*)H, (size_t)NN * NN / 4);
    diag_kernel<<<NN / 256, 256, 0, stream>>>(H);
    scatter_kernel<<<NN * TOPK / 256, 256, 0, stream>>>(H, tv, ti);
    colsum_kernel<<<NN / 256, 256, 0, stream>>>(H, cs);
    diagfix_kernel<<<NN / 256, 256, 0, stream>>>(H, cs);

    gemm_bias_kernel<<<dim3(NN / 32, 1), 128, 0, stream>>>(
        fusb, ew1b, ew_b1, hidb, 256, 1.0f, 1);
    ew_kernel<<<NN / 8, 256, 0, stream>>>(hidb, ew_w2, ew_b2, ew);
}
